// SA_global_40415642255797
// MI455X (gfx1250) — matrix operand order repaired and checked by reading, not compiled, not run on hardware
//
#include <hip/hip_runtime.h>

typedef __attribute__((ext_vector_type(16))) _Float16 v16h;
typedef __attribute__((ext_vector_type(8)))  _Float16 v8h;
typedef __attribute__((ext_vector_type(8)))  float    v8f;
typedef _Float16 half_t;

#define HW 16384  // 128*128 pixels per (b, channel)

static __device__ __forceinline__ v16h cat8(v8h lo, v8h hi) {
  return __builtin_shufflevector(lo, hi, 0,1,2,3,4,5,6,7,8,9,10,11,12,13,14,15);
}

// ---------------------------------------------------------------------------
// Kernel 1: QKV pointwise GEMM  (W[384x128] @ X[128 x 16384] per batch/branch)
// Scatter-stores Q,K (window layout, f16), V (window layout transposed, f16)
// and V spatial (f32, for LePE).
// grid = (512 pixel tiles, 3 qkv, 8 branch*batch), 256 threads (8 waves)
// ---------------------------------------------------------------------------
__global__ __launch_bounds__(256) void qkv_gemm_kernel(
    const float* __restrict__ x, const float* __restrict__ qkv_w,
    half_t* __restrict__ Qb, half_t* __restrict__ Kb, half_t* __restrict__ Vb,
    float* __restrict__ Vspat)
{
  __shared__ __align__(16) half_t Wl[128 * 136];  // [m][k] f16, padded
  __shared__ __align__(16) half_t Xt[32 * 136];   // [pixel][k] f16 (X^T), padded

  const int t  = threadIdx.x;
  const int pb = blockIdx.x;            // pixel tile (32 pixels)
  const int mb = blockIdx.y;            // 0=Q 1=K 2=V (128 out-channels each)
  const int bb = blockIdx.z;            // branch*4 + b
  const int branch = bb >> 2;
  const int pBase  = pb * 32;

  // Stage weights (f32 -> f16), rows mb*128 .. +128, all 128 cols
  const float* wsrc = qkv_w + (size_t)mb * 128 * 128;
  #pragma unroll
  for (int i = 0; i < 64; ++i) {
    int e = t + i * 256;
    Wl[(e >> 7) * 136 + (e & 127)] = (half_t)wsrc[e];
  }
  // Stage X transposed: 32 pixels x 128 input channels
  const float* xsrc = x + ((size_t)(bb & 3) * 256 + branch * 128) * HW + pBase;
  #pragma unroll
  for (int i = 0; i < 16; ++i) {
    int e = t + i * 256;
    int p = e & 31, c = e >> 5;
    Xt[p * 136 + c] = (half_t)xsrc[(size_t)c * HW + p];
  }
  __syncthreads();

  const int wv   = t >> 5;       // wave 0..7 -> 16 output rows each
  const int lane = t & 31;
  const int lm   = lane & 15;
  const int hf   = lane >> 4;

  v8f acc0 = {}; v8f acc1 = {};
  const int m = wv * 16 + lm;
  #pragma unroll
  for (int kc = 0; kc < 4; ++kc) {
    // A fragment: K(e) = kc*32 + 16*hf + e  (contiguous)
    const half_t* ap = &Wl[m * 136 + kc * 32 + hf * 16];
    v16h a = cat8(*(const v8h*)ap, *(const v8h*)(ap + 8));
    // B fragments: K(e) = kc*32 + 16*hf + e  (contiguous)
    v16h b0 = *(const v16h*)(&Xt[lm * 136 + kc * 32 + hf * 16]);
    v16h b1 = *(const v16h*)(&Xt[(16 + lm) * 136 + kc * 32 + hf * 16]);
    acc0 = __builtin_amdgcn_wmma_f32_16x16x32_f16(false, a, false, b0, (short)0, acc0, false, false);
    acc1 = __builtin_amdgcn_wmma_f32_16x16x32_f16(false, a, false, b1, (short)0, acc1, false, false);
  }

  const float qscale = 0.17677669529663687f;  // 32^-0.5, folded into Q
  #pragma unroll
  for (int nb = 0; nb < 2; ++nb) {
    v8f acc = nb ? acc1 : acc0;
    int p = pBase + nb * 16 + lm;
    int row = p >> 7, col = p & 127;
    int win, tok;
    if (branch == 0) { win = col >> 1; tok = row * 2 + (col & 1); }         // vertical strips
    else             { win = row >> 1; tok = ((row & 1) << 7) + col; }      // horizontal strips
    #pragma unroll
    for (int r = 0; r < 8; ++r) {
      int cp   = wv * 16 + r + 8 * hf;          // channel 0..127 within q/k/v
      int head = cp & 3, hd = cp >> 2;          // channel = hd*nh + head
      float val = acc[r];
      size_t hwbase = (((size_t)bb * 64 + win) * 4 + head);
      if (mb == 0) {
        Qb[(hwbase * 256 + tok) * 32 + hd] = (half_t)(val * qscale);
      } else if (mb == 1) {
        Kb[(hwbase * 256 + tok) * 32 + hd] = (half_t)val;
      } else {
        Vb[(hwbase * 32 + hd) * 256 + tok] = (half_t)val;   // transposed [hd][tok]
        Vspat[((size_t)bb * 128 + cp) * HW + p] = val;
      }
    }
  }
}

// ---------------------------------------------------------------------------
// Kernel 2: per head-window attention (256 tokens, hd=32) + LePE + store.
// grid = (4 heads, 64 windows, 8 branch*batch), 256 threads (8 waves).
// Each wave: 16 query rows per pass, 2 passes. S kept in registers (16 v8f).
// ---------------------------------------------------------------------------
__global__ __launch_bounds__(256) void attn_kernel(
    const half_t* __restrict__ Qb, const half_t* __restrict__ Kb,
    const half_t* __restrict__ Vb, const float* __restrict__ Vspat,
    const float* __restrict__ lepe_w, const float* __restrict__ lepe_b,
    float* __restrict__ out)
{
  __shared__ __align__(16) half_t Pl[8 * 16 * 40];  // per-wave P chunk, padded
  __shared__ float w3[128 * 9];
  __shared__ float bv[128];

  const int t    = threadIdx.x;
  const int head = blockIdx.x;
  const int win  = blockIdx.y;
  const int bb   = blockIdx.z;
  const int branch = bb >> 2;
  const int b      = bb & 3;

  if (t < 128) bv[t] = lepe_b[t];
  for (int e = t; e < 128 * 9; e += 256) w3[e] = lepe_w[e];
  __syncthreads();

  const size_t hw = ((size_t)bb * 64 + win) * 4 + head;
  const half_t* Qg = Qb + hw * 8192;   // [tok][hd]
  const half_t* Kg = Kb + hw * 8192;   // [tok][hd]
  const half_t* Vg = Vb + hw * 8192;   // [hd][tok]

  const int wv   = t >> 5;
  const int lane = t & 31;
  const int lm   = lane & 15;
  const int hf   = lane >> 4;
  half_t* Pw = &Pl[wv * 16 * 40];

  for (int g = 0; g < 2; ++g) {
    const int q0 = (wv + g * 8) * 16;  // 16 query rows for this wave/pass

    // Q A-fragment (hd = 32 = one K chunk), straight from global
    const half_t* qp = Qg + (size_t)(q0 + lm) * 32 + hf * 16;
    v16h qa = cat8(*(const v8h*)qp, *(const v8h*)(qp + 8));

    // S = Q * K^T over all 256 keys (16 accumulator blocks)
    v8f s[16];
    #pragma unroll
    for (int i = 0; i < 16; ++i) {
      v16h kb = *(const v16h*)(Kg + (size_t)(i * 16 + lm) * 32 + hf * 16);
      v8f z = {};
      s[i] = __builtin_amdgcn_wmma_f32_16x16x32_f16(false, qa, false, kb, (short)0, z, false, false);
    }

    // Row softmax: row M = r + 8*hf lives in 16 lanes (same hf), 16 blocks
    float rinv[8];
    #pragma unroll
    for (int r = 0; r < 8; ++r) {
      float m = -3.0e38f;
      #pragma unroll
      for (int i = 0; i < 16; ++i) m = fmaxf(m, s[i][r]);
      m = fmaxf(m, __shfl_xor(m, 1));
      m = fmaxf(m, __shfl_xor(m, 2));
      m = fmaxf(m, __shfl_xor(m, 4));
      m = fmaxf(m, __shfl_xor(m, 8));
      float ssum = 0.f;
      #pragma unroll
      for (int i = 0; i < 16; ++i) { float e = __expf(s[i][r] - m); s[i][r] = e; ssum += e; }
      ssum += __shfl_xor(ssum, 1);
      ssum += __shfl_xor(ssum, 2);
      ssum += __shfl_xor(ssum, 4);
      ssum += __shfl_xor(ssum, 8);
      rinv[r] = 1.0f / ssum;
    }

    // O = P * V, restaging P through per-wave LDS to get the A layout
    v8f o0 = {}, o1 = {};
    #pragma unroll
    for (int kc = 0; kc < 8; ++kc) {
      #pragma unroll
      for (int j = 0; j < 2; ++j) {
        #pragma unroll
        for (int r = 0; r < 8; ++r)
          Pw[(r + 8 * hf) * 40 + j * 16 + lm] = (half_t)(s[kc * 2 + j][r] * rinv[r]);
      }
      __syncthreads();  // uniform trip counts; also orders LDS write->read
      const half_t* pp = &Pw[lm * 40 + hf * 16];
      v16h pa = cat8(*(const v8h*)pp, *(const v8h*)(pp + 8));
      v16h v0 = *(const v16h*)(Vg + (size_t)lm * 256 + kc * 32 + hf * 16);
      v16h v1 = *(const v16h*)(Vg + (size_t)(16 + lm) * 256 + kc * 32 + hf * 16);
      o0 = __builtin_amdgcn_wmma_f32_16x16x32_f16(false, pa, false, v0, (short)0, o0, false, false);
      o1 = __builtin_amdgcn_wmma_f32_16x16x32_f16(false, pa, false, v1, (short)0, o1, false, false);
      __syncthreads();
    }

    // Add LePE (3x3 depthwise on spatial V) and store fp32 output
    #pragma unroll
    for (int nb = 0; nb < 2; ++nb) {
      v8f oc = nb ? o1 : o0;
      int nhd = nb * 16 + lm;              // hd channel of this column
      int cp  = nhd * 4 + head;            // channel within branch half
      const float* vsp = Vspat + ((size_t)bb * 128 + cp) * HW;
      const float* wk  = &w3[cp * 9];
      float bias = bv[cp];
      #pragma unroll
      for (int r = 0; r < 8; ++r) {
        int tok = q0 + r + 8 * hf;
        int row, col;
        if (branch == 0) { row = tok >> 1; col = win * 2 + (tok & 1); }
        else             { row = win * 2 + (tok >> 7); col = tok & 127; }
        float acc = bias;
        #pragma unroll
        for (int dy = -1; dy <= 1; ++dy) {
          int rr = row + dy;
          if (rr < 0 || rr >= 128) continue;
          #pragma unroll
          for (int dx = -1; dx <= 1; ++dx) {
            int cc = col + dx;
            if (cc < 0 || cc >= 128) continue;
            acc += wk[(dy + 1) * 3 + (dx + 1)] * vsp[rr * 128 + cc];
          }
        }
        out[((size_t)b * 256 + branch * 128 + cp) * HW + row * 128 + col] = oc[r] + acc;
      }
    }
  }
}

// ---------------------------------------------------------------------------
// Workspace: Q f16 32MiB | K f16 32MiB | V f16 32MiB | Vspat f32 64MiB
// ---------------------------------------------------------------------------
extern "C" void kernel_launch(void* const* d_in, const int* in_sizes, int n_in,
                              void* d_out, int out_size, void* d_ws, size_t ws_size,
                              hipStream_t stream) {
  const float* x      = (const float*)d_in[0];
  const float* qkv_w  = (const float*)d_in[1];
  const float* lepe_w = (const float*)d_in[2];
  const float* lepe_b = (const float*)d_in[3];
  float* out = (float*)d_out;

  char* ws = (char*)d_ws;
  half_t* Qb    = (half_t*)(ws);
  half_t* Kb    = (half_t*)(ws + ((size_t)32 << 20));
  half_t* Vb    = (half_t*)(ws + ((size_t)64 << 20));
  float*  Vspat = (float*) (ws + ((size_t)96 << 20));

  qkv_gemm_kernel<<<dim3(512, 3, 8), 256, 0, stream>>>(x, qkv_w, Qb, Kb, Vb, Vspat);
  attn_kernel<<<dim3(4, 64, 8), 256, 0, stream>>>(Qb, Kb, Vb, Vspat, lepe_w, lepe_b, out);
}
